// BiAttention_79173427134694
// MI455X (gfx1250) — matrix operand order repaired and checked by reading, not compiled, not run on hardware
//
#include <hip/hip_runtime.h>
#include <hip/hip_bf16.h>

// ---------------------------------------------------------------------------
// BiAttention for MI455X (gfx1250): all heavy math routed through
// v_wmma_f32_16x16x32_bf16 (wave32 WMMA, f32 accumulate).
//
// Pipeline (all on `stream`, in order):
//   1. amr_t  = amr  @ W_amr^T  + b_amr          (NT GEMM)
//   2. text_t = text @ W_text^T + b_text         (NT GEMM)
//   3. S_b    = amr_t_b @ text_t_b^T  -> d_out[amr_w region]   (NT, batched)
//   4. softmax over batch axis in place; text_w = transpose(amr_w)
//   5. att_text_b = amr_w_b  @ text_b   -> combined[:,768:]    (NN, batched)
//   6. att_amr_b  = text_w_b @ amr_t_b  -> combined[:,0:768]   (NN, batched)
//   7. out = combined @ W_out^T + b_out                        (NT GEMM)
//
// GEMM: 256 threads (8 wave32), block tile 128x128, K-step 32,
// wave tile 64x32 => 8 WMMAs per wave per barrier iteration.
// ---------------------------------------------------------------------------

typedef __attribute__((ext_vector_type(16))) __bf16 v16bf;
typedef __attribute__((ext_vector_type(8)))  __bf16 v8bf;
typedef __attribute__((ext_vector_type(4)))  __bf16 v4bf;
typedef __attribute__((ext_vector_type(8)))  float  v8f;

#define LDS_PAD 40  // 32 bf16 payload + 8 pad (row stride 80B, 16B aligned)

// A fragment: 16x32 bf16 tile, row-major in LDS.
// Layout: lane m (0-15) holds K=0..15; lane m+16 holds
// K=16..31 (two 16-byte chunks per lane).
__device__ __forceinline__ v16bf load_a_frag(const __bf16 lds[][LDS_PAD],
                                             int rbase, int lane) {
  const int r = rbase + (lane & 15);
  const int h = (lane >> 4) * 16;
  v8bf lo = *(const v8bf*)&lds[r][h];
  v8bf hi = *(const v8bf*)&lds[r][8 + h];
  return __builtin_shufflevector(lo, hi, 0, 1, 2, 3, 4, 5, 6, 7,
                                 8, 9, 10, 11, 12, 13, 14, 15);
}

// B fragment: B^T is 32x16; LDS holds B as [n][k] row-major.
// Lane n (0-15) holds K=0..15 of column n; lane n+16 holds K=16..31.
__device__ __forceinline__ v16bf load_b_frag(const __bf16 lds[][LDS_PAD],
                                             int nbase, int lane) {
  const int n = nbase + (lane & 15);
  const int h = (lane >> 4) * 16;
  return *(const v16bf*)&lds[n][h];
}

// Generic mixed-precision GEMM: C[M,N] = A[M,K] * op(B) (+bias), f32 in/out,
// bf16 WMMA inside.  B_NT: B stored [N][K] row-major (op = B^T).
//                   !B_NT: B stored [K][N] row-major (op = B).
template <bool B_NT>
__global__ __launch_bounds__(256) void gemm_bf16_wmma(
    const float* __restrict__ A, long lda, long strideAb,
    const float* __restrict__ B, long ldb, long strideBb,
    float* __restrict__ C, long ldc, long strideCb,
    const float* __restrict__ bias, int K) {
  __shared__ __bf16 lds_a[128][LDS_PAD];
  __shared__ __bf16 lds_b[128][LDS_PAD];

  const int t    = threadIdx.x;
  const int lane = t & 31;
  const int wid  = t >> 5;
  const int wm   = (wid & 1) * 64;   // 2 waves tile M (64 rows each)
  const int wn   = (wid >> 1) * 32;  // 4 waves tile N (32 cols each)

  const long row0 = (long)blockIdx.y * 128;
  const long col0 = (long)blockIdx.x * 128;
  const float* Ab = A + (long)blockIdx.z * strideAb;
  const float* Bb = B + (long)blockIdx.z * strideBb;
  float*       Cb = C + (long)blockIdx.z * strideCb;

  v8f acc[4][2] = {};

  for (int kt = 0; kt < K; kt += 32) {
    // ---- stage A tile: 128 rows x 32 k (f32 -> bf16) ----
    {
      const int col4 = t & 7;   // float4 index along K
      const int r0   = t >> 3;  // 0..31
#pragma unroll
      for (int i = 0; i < 4; ++i) {
        const int r = r0 + i * 32;
        const float4 f =
            *(const float4*)(Ab + (row0 + r) * lda + kt + col4 * 4);
        v4bf q = {(__bf16)f.x, (__bf16)f.y, (__bf16)f.z, (__bf16)f.w};
        *(v4bf*)&lds_a[r][col4 * 4] = q;
      }
    }
    // ---- stage B tile into [n][k] layout: 128 n x 32 k ----
    if (B_NT) {
      const int col4 = t & 7;
      const int n0   = t >> 3;  // 0..31
#pragma unroll
      for (int i = 0; i < 4; ++i) {
        const int n = n0 + i * 32;
        const float4 f =
            *(const float4*)(Bb + (col0 + n) * ldb + kt + col4 * 4);
        v4bf q = {(__bf16)f.x, (__bf16)f.y, (__bf16)f.z, (__bf16)f.w};
        *(v4bf*)&lds_b[n][col4 * 4] = q;
      }
    } else {
      const int n4  = t & 31;   // float4 over 128 n's
      const int kk0 = t >> 5;   // 0..7
#pragma unroll
      for (int i = 0; i < 4; ++i) {
        const int kk = kk0 + i * 8;
        const float4 f =
            *(const float4*)(Bb + (long)(kt + kk) * ldb + col0 + n4 * 4);
        lds_b[n4 * 4 + 0][kk] = (__bf16)f.x;
        lds_b[n4 * 4 + 1][kk] = (__bf16)f.y;
        lds_b[n4 * 4 + 2][kk] = (__bf16)f.z;
        lds_b[n4 * 4 + 3][kk] = (__bf16)f.w;
      }
    }
    __syncthreads();

    v16bf af[4], bf[2];
#pragma unroll
    for (int i = 0; i < 4; ++i) af[i] = load_a_frag(lds_a, wm + i * 16, lane);
#pragma unroll
    for (int j = 0; j < 2; ++j) bf[j] = load_b_frag(lds_b, wn + j * 16, lane);

#pragma unroll
    for (int i = 0; i < 4; ++i)
#pragma unroll
      for (int j = 0; j < 2; ++j)
        acc[i][j] = __builtin_amdgcn_wmma_f32_16x16x32_bf16(
            false, af[i], false, bf[j], (short)0, acc[i][j], false, false);
    __syncthreads();
  }

  // ---- store: f32 C/D layout (VGPR v -> row v / v+8 per lane half) ----
  const int cc    = lane & 15;
  const int rhalf = (lane >> 4) * 8;
#pragma unroll
  for (int i = 0; i < 4; ++i) {
#pragma unroll
    for (int j = 0; j < 2; ++j) {
      const long col = col0 + wn + j * 16 + cc;
      const float bv = bias ? bias[col] : 0.0f;
#pragma unroll
      for (int v = 0; v < 8; ++v) {
        const long row = row0 + wm + i * 16 + rhalf + v;
        Cb[row * ldc + col] = acc[i][j][v] + bv;
      }
    }
  }
}

// Softmax over the batch axis (B=8) of w[b][n][m] (in place), plus
// transposed copy wt[b][m][n] = w[b][n][m] through LDS for coalescing.
__global__ __launch_bounds__(1024) void softmax_over_batch(
    float* __restrict__ w, float* __restrict__ wt) {
  __shared__ float tile[32][33];
  const int tx = threadIdx.x, ty = threadIdx.y;
  const long m  = (long)blockIdx.x * 32 + tx;
  const long n  = (long)blockIdx.y * 32 + ty;
  const long bs = 2048L * 2048L;

  float v[8];
  float mx = -3.4e38f;
#pragma unroll
  for (int b = 0; b < 8; ++b) {
    v[b] = w[b * bs + n * 2048 + m];
    mx = fmaxf(mx, v[b]);
  }
  float s = 0.f;
#pragma unroll
  for (int b = 0; b < 8; ++b) {
    v[b] = __expf(v[b] - mx);
    s += v[b];
  }
  const float inv = 1.f / s;
#pragma unroll
  for (int b = 0; b < 8; ++b) {
    v[b] *= inv;
    w[b * bs + n * 2048 + m] = v[b];
  }
  const long mo = (long)blockIdx.x * 32;
  const long no = (long)blockIdx.y * 32;
  for (int b = 0; b < 8; ++b) {
    __syncthreads();
    tile[ty][tx] = v[b];
    __syncthreads();
    wt[b * bs + (mo + ty) * 2048 + (no + tx)] = tile[tx][ty];
  }
}

extern "C" void kernel_launch(void* const* d_in, const int* in_sizes, int n_in,
                              void* d_out, int out_size, void* d_ws,
                              size_t ws_size, hipStream_t stream) {
  const float* amr    = (const float*)d_in[0];  // [8,2048,1024]
  const float* text   = (const float*)d_in[1];  // [8,2048,768]
  const float* W_amr  = (const float*)d_in[2];  // [768,1024]
  const float* b_amr  = (const float*)d_in[3];  // [768]
  const float* W_text = (const float*)d_in[4];  // [768,768]
  const float* b_text = (const float*)d_in[5];  // [768]
  const float* W_out  = (const float*)d_in[6];  // [768,1536]
  const float* b_out  = (const float*)d_in[7];  // [768]

  constexpr long B = 8, L = 2048, AMR = 1024, TXT = 768, H = 768;
  const long ML = B * L;  // 16384 fused batch*seq rows

  float* out = (float*)d_out;
  float* outp   = out;                 // [8,2048,768]
  float* amr_w  = out + ML * H;        // [8,2048,2048]
  float* text_w = amr_w + B * L * L;   // [8,2048,2048]

  // workspace: amr_t | combined (text_t aliases combined's first half;
  // text_t is dead before combined is written).  Peak = 151 MB.
  float* ws       = (float*)d_ws;
  float* amr_t    = ws;                // [16384,768]
  float* combined = ws + ML * H;       // [16384,1536]
  float* text_t   = combined;          // [16384,768] (aliased)

  const dim3 blk(256);

  // 1) amr_t = amr @ W_amr^T + b_amr        (NT, K=1024)
  gemm_bf16_wmma<true><<<dim3(H / 128, ML / 128, 1), blk, 0, stream>>>(
      amr, AMR, 0, W_amr, AMR, 0, amr_t, H, 0, b_amr, (int)AMR);

  // 2) text_t = text @ W_text^T + b_text    (NT, K=768)
  gemm_bf16_wmma<true><<<dim3(H / 128, ML / 128, 1), blk, 0, stream>>>(
      text, TXT, 0, W_text, TXT, 0, text_t, H, 0, b_text, (int)TXT);

  // 3) scores S_b = amr_t_b @ text_t_b^T -> amr_w region (NT, batched, K=768)
  gemm_bf16_wmma<true><<<dim3(L / 128, L / 128, B), blk, 0, stream>>>(
      amr_t, H, L * H, text_t, H, L * H, amr_w, L, L * L, nullptr, (int)H);

  // 4) softmax over batch axis; text_w = transpose(amr_w)
  softmax_over_batch<<<dim3(L / 32, L / 32), dim3(32, 32), 0, stream>>>(
      amr_w, text_w);

  // 5) att_text_b = amr_w_b @ text_b -> combined[:,768:] (NN, batched, K=2048)
  gemm_bf16_wmma<false><<<dim3(TXT / 128, L / 128, B), blk, 0, stream>>>(
      amr_w, L, L * L, text, TXT, L * TXT, combined + H, 2 * H, L * 2 * H,
      nullptr, (int)L);

  // 6) att_amr_b = text_w_b @ amr_t_b -> combined[:,0:768] (NN, batched, K=2048)
  gemm_bf16_wmma<false><<<dim3(H / 128, L / 128, B), blk, 0, stream>>>(
      text_w, L, L * L, amr_t, H, L * H, combined, 2 * H, L * 2 * H,
      nullptr, (int)L);

  // 7) out = combined @ W_out^T + b_out     (NT, K=1536)
  gemm_bf16_wmma<true><<<dim3(H / 128, ML / 128, 1), blk, 0, stream>>>(
      combined, 2 * H, 0, W_out, 2 * H, 0, outp, H, 0, b_out, (int)(2 * H));
}
